// Attention_35579509080675
// MI455X (gfx1250) — compile-verified, matrix operand order checked by reading, not run on hardware
//
#include <hip/hip_runtime.h>

typedef __attribute__((ext_vector_type(16))) __bf16 v16bf;
typedef __attribute__((ext_vector_type(8)))  float  v8f;
typedef int v4i_vs __attribute__((vector_size(4 * sizeof(int))));

#define B_  2
#define S_  2048
#define D_  2048
#define H_  32
#define HD_ 64
#define BS_ (B_ * S_)

#if defined(__has_builtin)
#if __has_builtin(__builtin_amdgcn_global_load_async_to_lds_b128)
#define HAS_ASYNC_LDS 1
#endif
#endif
#ifndef HAS_ASYNC_LDS
#define HAS_ASYNC_LDS 0
#endif

// Builtin expects: (int4 AS1* src, int4 AS3* dst, imm offset, imm cpol)
#define GPTR4(p) ((__attribute__((address_space(1))) v4i_vs*)(void*)(p))
#define LPTR4(p) ((__attribute__((address_space(3))) v4i_vs*)(void*)(p))

// k-index pattern for 16-bit A/B WMMA fragments (wave32):
// lanes 0-15 hold K in {0..7, 16..23}; lanes 16-31 hold K in {8..15, 24..31}
__device__ __forceinline__ int kpat(int e, int hi) {
    return e + ((e >> 3) << 3) + (hi << 3);
}

// Copy 32 bytes (16 bf16) global -> LDS. Async path uses two b128 async ops
// (16 B per lane each), second one via the instruction immediate offset.
__device__ __forceinline__ void cp32_g2l(const __bf16* __restrict__ g, __bf16* l) {
#if HAS_ASYNC_LDS
    __builtin_amdgcn_global_load_async_to_lds_b128(GPTR4(g), LPTR4(l), 0, 0);
    __builtin_amdgcn_global_load_async_to_lds_b128(GPTR4(g), LPTR4(l), 16, 0);
#else
    *(float4*)l       = *(const float4*)g;
    *(float4*)(l + 8) = *(const float4*)(g + 8);
#endif
}

__device__ __forceinline__ void wait_async_lds() {
#if HAS_ASYNC_LDS
#if __has_builtin(__builtin_amdgcn_s_wait_asynccnt)
    __builtin_amdgcn_s_wait_asynccnt(0);
#else
    asm volatile("s_wait_asynccnt 0x0" ::: "memory");
#endif
#endif
}

// ---------------------------------------------------------------------------
// fp32 -> bf16 conversion
// ---------------------------------------------------------------------------
__global__ void f32_to_bf16(const float* __restrict__ in, __bf16* __restrict__ out, int n) {
    int i = blockIdx.x * blockDim.x + threadIdx.x;
    if (i < n) out[i] = (__bf16)in[i];
}

// ---------------------------------------------------------------------------
// Tiled bf16 GEMM: C[M,N] = A[M,K] * B[K,N], row-major, WMMA 16x16x32 bf16.
// 256 threads = 8 waves; block tile 128x128; each wave does 64x32 (8 accums).
// Double-buffered LDS with async global->LDS copies.
// ---------------------------------------------------------------------------
template <bool OUT_F32>
__global__ __launch_bounds__(256) void gemm_wmma(const __bf16* __restrict__ A,
                                                 const __bf16* __restrict__ Bm,
                                                 void* __restrict__ Cp,
                                                 int M, int N, int K) {
    __shared__ __bf16 As[2][128][40];   // 80B row stride: 16B aligned, conflict-free
    __shared__ __bf16 Bs[2][32][136];   // 272B row stride: 16B aligned

    const int t    = threadIdx.x;
    const int lane = t & 31;
    const int w    = t >> 5;
    const int hi   = lane >> 4;
    const int ln   = lane & 15;
    const int m0   = blockIdx.y * 128;
    const int n0   = blockIdx.x * 128;
    const int wm   = (w >> 2) * 64;  // 0 or 64
    const int wn   = (w & 3) * 32;   // 0,32,64,96

    const int ar = t >> 1, ac = (t & 1) * 16;   // A tile: 128 rows x 32 cols
    const int br = t >> 3, bc = (t & 7) * 16;   // B tile: 32 rows x 128 cols

    const __bf16* gA = A  + (size_t)(m0 + ar) * K + ac;
    const __bf16* gB = Bm + (size_t)br * N + n0 + bc;

    v8f acc[4][2] = {};

    // prologue: fill buffer 0
    cp32_g2l(gA, &As[0][ar][ac]);
    cp32_g2l(gB, &Bs[0][br][bc]);
    wait_async_lds();
    __syncthreads();

    for (int k0 = 0; k0 < K; k0 += 32) {
        const int cur = (k0 >> 5) & 1;
        if (k0 + 32 < K) {   // issue async copies for next tile into other buffer
            cp32_g2l(gA + k0 + 32, &As[cur ^ 1][ar][ac]);
            cp32_g2l(gB + (size_t)(k0 + 32) * N, &Bs[cur ^ 1][br][bc]);
        }

        v16bf af[4], bfv[2];
#pragma unroll
        for (int a = 0; a < 4; ++a)
#pragma unroll
            for (int e = 0; e < 16; ++e)
                af[a][e] = As[cur][wm + a * 16 + ln][kpat(e, hi)];
#pragma unroll
        for (int b = 0; b < 2; ++b)
#pragma unroll
            for (int e = 0; e < 16; ++e)
                bfv[b][e] = Bs[cur][kpat(e, hi)][wn + b * 16 + ln];

#pragma unroll
        for (int a = 0; a < 4; ++a)
#pragma unroll
            for (int b = 0; b < 2; ++b)
                acc[a][b] = __builtin_amdgcn_wmma_f32_16x16x32_bf16(
                    false, af[a], false, bfv[b], (short)0, acc[a][b], false, false);

        wait_async_lds();
        __syncthreads();
    }

#pragma unroll
    for (int a = 0; a < 4; ++a)
#pragma unroll
        for (int b = 0; b < 2; ++b)
#pragma unroll
            for (int i = 0; i < 8; ++i) {
                const int row = m0 + wm + a * 16 + i + hi * 8;
                const int col = n0 + wn + b * 16 + ln;
                if (OUT_F32)
                    ((float*)Cp)[(size_t)row * N + col] = acc[a][b][i];
                else
                    ((__bf16*)Cp)[(size_t)row * N + col] = (__bf16)acc[a][b][i];
            }
}

// ---------------------------------------------------------------------------
// RoPE, in-place on bf16 tensor laid out (B*S, H*HD). One thread per pair.
// ---------------------------------------------------------------------------
__global__ void rope_kernel(__bf16* __restrict__ qk,
                            const float* __restrict__ cosf,
                            const float* __restrict__ sinf) {
    int idx = blockIdx.x * blockDim.x + threadIdx.x;   // B*S*H*(HD/2)
    if (idx >= BS_ * H_ * (HD_ / 2)) return;
    const int i   = idx & 31;          // pair index within head
    const int h   = (idx >> 5) & 31;   // head
    const int row = idx >> 10;         // b*S + s
    const int s   = row & (S_ - 1);
    const float c  = cosf[s * 32 + i];
    const float sn = sinf[s * 32 + i];
    const size_t off = (size_t)row * D_ + h * HD_ + 2 * i;
    const float tr = (float)qk[off];
    const float ti = (float)qk[off + 1];
    qk[off]     = (__bf16)(tr * c - ti * sn);
    qk[off + 1] = (__bf16)(tr * sn + ti * c);
}

// ---------------------------------------------------------------------------
// Flash attention (causal). Grid (S/64, H, B), 128 threads = 4 waves.
// Each wave handles 16 query rows; K/V staged 32 keys at a time in LDS,
// double-buffered with async global->LDS copies.
// ---------------------------------------------------------------------------
__global__ __launch_bounds__(128) void flash_attn(const __bf16* __restrict__ Q,
                                                  const __bf16* __restrict__ Kt,
                                                  const __bf16* __restrict__ V,
                                                  __bf16* __restrict__ O) {
    __shared__ __bf16 lds_k[2][32][72];
    __shared__ __bf16 lds_v[2][32][72];
    __shared__ __bf16 lds_p[4][16][40];

    const int t    = threadIdx.x;
    const int lane = t & 31;
    const int w    = t >> 5;
    const int hi   = lane >> 4;
    const int ln   = lane & 15;
    const int q0   = blockIdx.x * 64;
    const int h    = blockIdx.y;
    const int b    = blockIdx.z;
    const int qrow = q0 + w * 16;
    const size_t base = (size_t)b * S_ * D_ + (size_t)h * HD_;

    // Resident Q fragments: 16 rows x 64 (two k-steps of 32)
    v16bf qf[2];
#pragma unroll
    for (int step = 0; step < 2; ++step)
#pragma unroll
        for (int e = 0; e < 16; ++e)
            qf[step][e] = Q[base + (size_t)(qrow + ln) * D_ + step * 32 + kpat(e, hi)];

    v8f o[4] = {};
    float mrun[8], lrun[8];
#pragma unroll
    for (int i = 0; i < 8; ++i) { mrun[i] = -3.0e38f; lrun[i] = 0.0f; }

    const int kr = t >> 2, kc = (t & 3) * 16;   // 32 keys x 64 dims, 32B/thread
    const float scale = 0.125f;                 // 1/sqrt(64)
    const int kend = q0 + 64;

    const __bf16* gK = Kt + base + (size_t)kr * D_ + kc;
    const __bf16* gV = V  + base + (size_t)kr * D_ + kc;

    // prologue: fill buffer 0 with key block 0
    cp32_g2l(gK, &lds_k[0][kr][kc]);
    cp32_g2l(gV, &lds_v[0][kr][kc]);
    wait_async_lds();
    __syncthreads();

    for (int k0 = 0; k0 < kend; k0 += 32) {
        const int cur = (k0 >> 5) & 1;
        if (k0 + 32 < kend) {
            cp32_g2l(gK + (size_t)(k0 + 32) * D_, &lds_k[cur ^ 1][kr][kc]);
            cp32_g2l(gV + (size_t)(k0 + 32) * D_, &lds_v[cur ^ 1][kr][kc]);
            if (k0 + 64 < kend)   // L2 warm-up for block after next
                __builtin_prefetch(gK + (size_t)(k0 + 64) * D_, 0, 0);
        }

        if (k0 <= qrow + 15) {     // wave-uniform causal skip
            // ---- scores: 16 queries x 32 keys ----
            v8f sc[2] = {};
#pragma unroll
            for (int step = 0; step < 2; ++step) {
                v16bf kf0, kf1;
#pragma unroll
                for (int e = 0; e < 16; ++e) {
                    kf0[e] = lds_k[cur][ln][step * 32 + kpat(e, hi)];
                    kf1[e] = lds_k[cur][16 + ln][step * 32 + kpat(e, hi)];
                }
                sc[0] = __builtin_amdgcn_wmma_f32_16x16x32_bf16(
                    false, qf[step], false, kf0, (short)0, sc[0], false, false);
                sc[1] = __builtin_amdgcn_wmma_f32_16x16x32_bf16(
                    false, qf[step], false, kf1, (short)0, sc[1], false, false);
            }

            // ---- scale + causal mask + online softmax ----
            float mr[8], ls[8];
#pragma unroll
            for (int i = 0; i < 8; ++i) {
                const int qi = qrow + i + hi * 8;
#pragma unroll
                for (int nt = 0; nt < 2; ++nt) {
                    const int key = k0 + nt * 16 + ln;
                    sc[nt][i] = sc[nt][i] * scale + (key > qi ? -1.0e9f : 0.0f);
                }
                mr[i] = fmaxf(sc[0][i], sc[1][i]);
            }
#pragma unroll
            for (int off = 1; off < 16; off <<= 1)
#pragma unroll
                for (int i = 0; i < 8; ++i)
                    mr[i] = fmaxf(mr[i], __shfl_xor(mr[i], off, 32));

            float alpha[8];
#pragma unroll
            for (int i = 0; i < 8; ++i) {
                const float mnew = fmaxf(mrun[i], mr[i]);
                alpha[i] = __expf(mrun[i] - mnew);
                mrun[i]  = mnew;
                const float p0 = __expf(sc[0][i] - mnew);
                const float p1 = __expf(sc[1][i] - mnew);
                sc[0][i] = p0; sc[1][i] = p1;
                ls[i] = p0 + p1;
            }
#pragma unroll
            for (int off = 1; off < 16; off <<= 1)
#pragma unroll
                for (int i = 0; i < 8; ++i)
                    ls[i] += __shfl_xor(ls[i], off, 32);
#pragma unroll
            for (int i = 0; i < 8; ++i)
                lrun[i] = lrun[i] * alpha[i] + ls[i];

            // ---- P (C layout) -> A fragment via per-wave LDS scratch ----
#pragma unroll
            for (int nt = 0; nt < 2; ++nt)
#pragma unroll
                for (int i = 0; i < 8; ++i)
                    lds_p[w][i + hi * 8][nt * 16 + ln] = (__bf16)sc[nt][i];
            asm volatile("s_wait_dscnt 0x0" ::: "memory");

            v16bf pf;
#pragma unroll
            for (int e = 0; e < 16; ++e)
                pf[e] = lds_p[w][ln][kpat(e, hi)];

            // ---- rescale O, then O += P * V ----
#pragma unroll
            for (int tt = 0; tt < 4; ++tt)
#pragma unroll
                for (int i = 0; i < 8; ++i)
                    o[tt][i] *= alpha[i];

#pragma unroll
            for (int tt = 0; tt < 4; ++tt) {
                v16bf vf;
#pragma unroll
                for (int e = 0; e < 16; ++e)
                    vf[e] = lds_v[cur][kpat(e, hi)][tt * 16 + ln];
                o[tt] = __builtin_amdgcn_wmma_f32_16x16x32_bf16(
                    false, pf, false, vf, (short)0, o[tt], false, false);
            }
        }

        wait_async_lds();
        __syncthreads();
    }

    // ---- normalize and store bf16 ----
    float inv[8];
#pragma unroll
    for (int i = 0; i < 8; ++i) inv[i] = 1.0f / lrun[i];
#pragma unroll
    for (int tt = 0; tt < 4; ++tt)
#pragma unroll
        for (int i = 0; i < 8; ++i) {
            const int row = qrow + i + hi * 8;
            O[base + (size_t)row * D_ + tt * 16 + ln] = (__bf16)(o[tt][i] * inv[i]);
        }
}

// ---------------------------------------------------------------------------
// Host side
// ---------------------------------------------------------------------------
extern "C" void kernel_launch(void* const* d_in, const int* in_sizes, int n_in,
                              void* d_out, int out_size, void* d_ws, size_t ws_size,
                              hipStream_t stream) {
    const float* x    = (const float*)d_in[0];
    const float* wq   = (const float*)d_in[1];
    const float* wk   = (const float*)d_in[2];
    const float* wv   = (const float*)d_in[3];
    const float* wo   = (const float*)d_in[4];
    const float* cosf = (const float*)d_in[5];
    const float* sinf = (const float*)d_in[6];
    (void)n_in; (void)in_sizes; (void)out_size; (void)ws_size;  // mask d_in[7] unused

    char* ws = (char*)d_ws;
    size_t off = 0;
    auto alloc = [&](size_t bytes) -> void* {
        void* p = ws + off;
        off += (bytes + 255) & ~(size_t)255;
        return p;
    };
    __bf16* xb  = (__bf16*)alloc((size_t)BS_ * D_ * 2);
    __bf16* wqb = (__bf16*)alloc((size_t)D_ * D_ * 2);
    __bf16* wkb = (__bf16*)alloc((size_t)D_ * D_ * 2);
    __bf16* wvb = (__bf16*)alloc((size_t)D_ * D_ * 2);
    __bf16* wob = (__bf16*)alloc((size_t)D_ * D_ * 2);
    __bf16* qb  = (__bf16*)alloc((size_t)BS_ * D_ * 2);
    __bf16* kb  = (__bf16*)alloc((size_t)BS_ * D_ * 2);
    __bf16* vb  = (__bf16*)alloc((size_t)BS_ * D_ * 2);
    __bf16* ab  = (__bf16*)alloc((size_t)BS_ * D_ * 2);

    const int nX = BS_ * D_;     // 8388608
    const int nW = D_ * D_;      // 4194304
    f32_to_bf16<<<nX / 256, 256, 0, stream>>>(x,  xb,  nX);
    f32_to_bf16<<<nW / 256, 256, 0, stream>>>(wq, wqb, nW);
    f32_to_bf16<<<nW / 256, 256, 0, stream>>>(wk, wkb, nW);
    f32_to_bf16<<<nW / 256, 256, 0, stream>>>(wv, wvb, nW);
    f32_to_bf16<<<nW / 256, 256, 0, stream>>>(wo, wob, nW);

    dim3 gg(D_ / 128, BS_ / 128);   // (16, 32)
    gemm_wmma<false><<<gg, 256, 0, stream>>>(xb, wqb, qb, BS_, D_, D_);
    gemm_wmma<false><<<gg, 256, 0, stream>>>(xb, wkb, kb, BS_, D_, D_);
    gemm_wmma<false><<<gg, 256, 0, stream>>>(xb, wvb, vb, BS_, D_, D_);

    const int nR = BS_ * H_ * (HD_ / 2);   // 4194304
    rope_kernel<<<nR / 256, 256, 0, stream>>>(qb, cosf, sinf);
    rope_kernel<<<nR / 256, 256, 0, stream>>>(kb, cosf, sinf);

    dim3 fg(S_ / 64, H_, B_);   // (32, 32, 2)
    flash_attn<<<fg, 128, 0, stream>>>(qb, kb, vb, ab);

    gemm_wmma<true><<<gg, 256, 0, stream>>>(ab, wob, d_out, BS_, D_, D_);
}
